// PatchCore_88235808129398
// MI455X (gfx1250) — matrix operand order repaired and checked by reading, not compiled, not run on hardware
//
#include <hip/hip_runtime.h>

typedef __attribute__((ext_vector_type(16))) __bf16 v16bf;
typedef __attribute__((ext_vector_type(8)))  float  v8f;

#define B_     8
#define C2     512
#define C3     1024
#define H0     28
#define W0     28
#define NPATCH (H0*W0)        // 784
#define NQ     (B_*NPATCH)    // 6272
#define EMB    1024
#define NBANK  16384

static __device__ __forceinline__ unsigned short f2bf(float f) {
  unsigned int u = __float_as_uint(f);
  u += 0x7FFFu + ((u >> 16) & 1u);          // round-to-nearest-even
  return (unsigned short)(u >> 16);
}
static __device__ __forceinline__ unsigned int enc_min(float f) {
  unsigned int b = __float_as_uint(f);
  return (b & 0x80000000u) ? ~b : (b | 0x80000000u);   // order-preserving uint key
}
static __device__ __forceinline__ float dec_min(unsigned int k) {
  return __uint_as_float((k & 0x80000000u) ? (k & 0x7FFFFFFFu) : ~k);
}

__global__ void init_kernel(unsigned int* __restrict__ minbuf) {
  int i = blockIdx.x * 256 + threadIdx.x;
  if (i < NQ) minbuf[i] = 0xFF800000u;      // enc_min(+inf)
}

// memory_bank f32 -> bf16 + squared row norms. One block per bank row.
__global__ void bank_kernel(const float* __restrict__ mb,
                            unsigned short* __restrict__ mb_bf,
                            float* __restrict__ mnorm) {
  __shared__ float red[256];
  int r = blockIdx.x, t = threadIdx.x;
  float4 v = ((const float4*)(mb + (size_t)r * EMB))[t];
  unsigned short* o = mb_bf + (size_t)r * EMB + t * 4;
  o[0] = f2bf(v.x); o[1] = f2bf(v.y); o[2] = f2bf(v.z); o[3] = f2bf(v.w);
  red[t] = v.x*v.x + v.y*v.y + v.z*v.z + v.w*v.w;
  __syncthreads();
  for (int off = 128; off > 0; off >>= 1) { if (t < off) red[t] += red[t + off]; __syncthreads(); }
  if (t == 0) mnorm[r] = red[0];
}

// 3x3 zero-padded box MEAN of feat3 -> s3[b][c][14][14]
__global__ void s3_kernel(const float* __restrict__ f3, float* __restrict__ s3) {
  int idx = blockIdx.x * 256 + threadIdx.x;           // 8*1024*196 total, exact
  int xx = idx % 14, yy = (idx / 14) % 14, bc = idx / 196;
  const float* p = f3 + (size_t)bc * 196;
  float s = 0.f;
  for (int dy = -1; dy <= 1; ++dy) {
    int ys = yy + dy; if ((unsigned)ys >= 14u) continue;
    for (int dx = -1; dx <= 1; ++dx) {
      int xs = xx + dx; if ((unsigned)xs >= 14u) continue;
      s += p[ys * 14 + xs];
    }
  }
  s3[idx] = s * (1.f / 9.f);
}

// Fused embedding: q[n][1024] (bf16) + ||q||^2. One block per patch n.
__global__ void q_kernel(const float* __restrict__ f2, const float* __restrict__ s3,
                         unsigned short* __restrict__ qbf, float* __restrict__ qnorm) {
  __shared__ float red[256];
  int n = blockIdx.x, t = threadIdx.x;
  int b = n / NPATCH, hw = n % NPATCH, y = hw / W0, x = hw % W0;
  // bilinear (half-pixel centers, edge clamp), 14 -> 28
  float cy = 0.5f * y - 0.25f, cx = 0.5f * x - 0.25f;
  int iy = (int)floorf(cy), ix = (int)floorf(cx);
  float wy = cy - iy, wx = cx - ix;
  int y0 = min(max(iy, 0), 13),     y1 = min(max(iy + 1, 0), 13);
  int x0 = min(max(ix, 0), 13),     x1 = min(max(ix + 1, 0), 13);
  float w00 = (1.f-wy)*(1.f-wx), w01 = (1.f-wy)*wx, w10 = wy*(1.f-wx), w11 = wy*wx;
  float acc = 0.f;
  for (int k = 0; k < 4; ++k) {
    int j = t + k * 256;
    float qv;
    if (j < C2) {                         // whole wave uniform: j<512 iff k<2
      const float* p = f2 + ((size_t)b * C2 + j) * (H0 * W0);
      float s = 0.f;
      for (int dy = -1; dy <= 1; ++dy) {
        int ys = y + dy; if ((unsigned)ys >= (unsigned)H0) continue;
        for (int dx = -1; dx <= 1; ++dx) {
          int xs = x + dx; if ((unsigned)xs >= (unsigned)W0) continue;
          s += p[ys * W0 + xs];
        }
      }
      qv = (s + p[y * W0 + x]) * 0.1f;    // (sum9 + center)/10
    } else {
      int c0 = 2 * (j - C2);
      const float* p0 = s3 + ((size_t)b * C3 + c0) * 196;
      const float* p1 = p0 + 196;
      float v0 = w00*p0[y0*14+x0] + w01*p0[y0*14+x1] + w10*p0[y1*14+x0] + w11*p0[y1*14+x1];
      float v1 = w00*p1[y0*14+x0] + w01*p1[y0*14+x1] + w10*p1[y1*14+x0] + w11*p1[y1*14+x1];
      qv = 0.5f * (v0 + v1);
    }
    qbf[(size_t)n * EMB + j] = f2bf(qv);
    acc += qv * qv;
  }
  red[t] = acc; __syncthreads();
  for (int off = 128; off > 0; off >>= 1) { if (t < off) red[t] += red[t + off]; __syncthreads(); }
  if (t == 0) qnorm[n] = red[0];
}

// WMMA bf16 GEMM + running min of squared L2 distance.
// grid = (NQ/16, NBANK/2048), 256 threads (8 waves); wave w owns 16x16 tiles.
__global__ __launch_bounds__(256) void nn_kernel(
    const unsigned short* __restrict__ qbf, const unsigned short* __restrict__ mbbf,
    const float* __restrict__ qnorm, const float* __restrict__ mnorm,
    unsigned int* __restrict__ minbuf) {
  __shared__ __align__(16) unsigned short qt[16 * EMB];   // 32 KB A tile
  int tid = threadIdx.x;
  int m0 = blockIdx.x * 16;
  int nbase = blockIdx.y * 2048;
  {
    const uint4* src = (const uint4*)(qbf + (size_t)m0 * EMB);
    uint4* dst = (uint4*)qt;
    for (int i = tid; i < 16 * EMB / 8; i += 256) dst[i] = src[i];
  }
  __syncthreads();
  int wave = tid >> 5, lane = tid & 31;
  int half = lane >> 4, l15 = lane & 15;
  float qn[8], vmin[8];
  for (int g = 0; g < 8; ++g) { qn[g] = qnorm[m0 + half * 8 + g]; vmin[g] = 3.4e38f; }
  // A layout (16x32 bf16): lanes0-15 M=lane K{0..7,16..23}; lanes16-31 K{8..15,24..31}
  const unsigned short* aBase = qt + l15 * EMB + half * 8;
  for (int it = 0; it < 16; ++it) {
    int n0 = nbase + (it * 8 + wave) * 16;
    // B layout (32x16 bf16): lane = column (bank row), K{0..7,16..23} / K{8..15,24..31} per half
    const unsigned short* bBase = mbbf + (size_t)(n0 + l15) * EMB + half * 8;
    if (it + 1 < 16)
      __builtin_prefetch(mbbf + (size_t)(nbase + ((it + 1) * 8 + wave) * 16 + l15) * EMB, 0, 1);
    v8f acc = {0.f, 0.f, 0.f, 0.f, 0.f, 0.f, 0.f, 0.f};
    for (int kb = 0; kb < EMB; kb += 32) {
      union { uint4 u[2]; v16bf v; } a, bb;
      a.u[0]  = *(const uint4*)(aBase + kb);        // K kb+ksel .. +7
      a.u[1]  = *(const uint4*)(aBase + kb + 16);   // K kb+16+ksel .. +7
      bb.u[0] = *(const uint4*)(bBase + kb);        // K kb+ksel .. +7
      bb.u[1] = *(const uint4*)(bBase + kb + 16);   // K kb+16+ksel .. +7
      acc = __builtin_amdgcn_wmma_f32_16x16x32_bf16(
          false, a.v, false, bb.v, (short)0, acc, false, false);
    }
    float mn = mnorm[n0 + l15];
    for (int g = 0; g < 8; ++g) {
      float d2 = qn[g] + mn - 2.f * acc[g];   // C/D: vgpr g = row m0+half*8+g, col n0+l15
      vmin[g] = fminf(vmin[g], d2);
    }
  }
  for (int g = 0; g < 8; ++g) {               // min over the 16 columns of each half
    float v = vmin[g];
    v = fminf(v, __shfl_xor(v, 1));
    v = fminf(v, __shfl_xor(v, 2));
    v = fminf(v, __shfl_xor(v, 4));
    v = fminf(v, __shfl_xor(v, 8));
    vmin[g] = v;
  }
  if (l15 == 0)
    for (int g = 0; g < 8; ++g)
      atomicMin(minbuf + m0 + half * 8 + g, enc_min(vmin[g]));
}

// decode mins -> patch scores, max-reduce -> image scores. One block per image.
__global__ void score_kernel(const unsigned int* __restrict__ minbuf, float* __restrict__ out) {
  __shared__ float red[256];
  int b = blockIdx.x, t = threadIdx.x;
  float m = -3.4e38f;
  for (int i = t; i < NPATCH; i += 256) {
    float f = dec_min(minbuf[b * NPATCH + i]);
    out[8 + b * NPATCH + i] = f;
    m = fmaxf(m, f);
  }
  red[t] = m; __syncthreads();
  for (int off = 128; off > 0; off >>= 1) { if (t < off) red[t] = fmaxf(red[t], red[t + off]); __syncthreads(); }
  if (t == 0) out[b] = red[0];
}

extern "C" void kernel_launch(void* const* d_in, const int* in_sizes, int n_in,
                              void* d_out, int out_size, void* d_ws, size_t ws_size,
                              hipStream_t stream) {
  (void)in_sizes; (void)n_in; (void)out_size; (void)ws_size;
  const float* feat2 = (const float*)d_in[0];   // [8,512,28,28]
  const float* feat3 = (const float*)d_in[1];   // [8,1024,14,14]
  const float* mbank = (const float*)d_in[2];   // [16384,1024]
  float* out = (float*)d_out;                   // [8] + [8,28,28]
  char* ws = (char*)d_ws;
  unsigned short* mb_bf = (unsigned short*)(ws);                 // 33,554,432 B
  unsigned short* q_bf  = (unsigned short*)(ws + 33554432);      // 12,845,056 B
  float* s3             = (float*)(ws + 46399488);               //  6,422,528 B
  float* mnorm          = (float*)(ws + 52822016);               //     65,536 B
  float* qnorm          = (float*)(ws + 52887552);               //     25,088 B
  unsigned int* minbuf  = (unsigned int*)(ws + 52912640);        //     25,088 B

  init_kernel <<<(NQ + 255) / 256, 256, 0, stream>>>(minbuf);
  bank_kernel <<<NBANK, 256, 0, stream>>>(mbank, mb_bf, mnorm);
  s3_kernel   <<<(B_ * C3 * 196) / 256, 256, 0, stream>>>(feat3, s3);
  q_kernel    <<<NQ, 256, 0, stream>>>(feat2, s3, q_bf, qnorm);
  nn_kernel   <<<dim3(NQ / 16, NBANK / 2048), 256, 0, stream>>>(q_bf, mb_bf, qnorm, mnorm, minbuf);
  score_kernel<<<B_, 256, 0, stream>>>(minbuf, out);
}
